// GNN_27934467293570
// MI455X (gfx1250) — matrix operand order repaired and checked by reading, not compiled, not run on hardware
//
#include <hip/hip_runtime.h>
#include <hip/hip_bf16.h>
#include <math.h>

#define N_NODES 50000
#define N_EDGES 800000
#define IN_DIM  128
#define H_DIM   128
#define OUT_DIM 112
#define EDGE_DIM 8
#define NEG_SLOPE 0.2f
#define BN_EPS 1e-5f

typedef __attribute__((ext_vector_type(16))) __bf16 v16bf;
typedef __attribute__((ext_vector_type(8)))  float  v8f;

union BfVec { v16bf v; unsigned short s[16]; uint4 q[2]; };

__device__ __forceinline__ unsigned short f32_bf16_rne(float f) {
    unsigned u = __float_as_uint(f);
    u += 0x7FFFu + ((u >> 16) & 1u);
    return (unsigned short)(u >> 16);
}
__device__ __forceinline__ float bf16_hi_f32(unsigned short s) {
    return __uint_as_float(((unsigned)s) << 16);
}
__device__ __forceinline__ float wave_sum(float v) {
#pragma unroll
    for (int off = 16; off; off >>= 1) v += __shfl_down(v, off, 32);
    return v;
}
__device__ __forceinline__ void atomicMaxF(float* addr, float v) {
    if (v >= 0.f) atomicMax((int*)addr, __float_as_int(v));
    else          atomicMin((unsigned int*)addr, __float_as_uint(v));
}

// ---------------------------------------------------------------------------
// Pre-pack W[K x Cout] into WMMA B-operand lane order, split into bf16 hi/lo.
// Packed layout: for (nTile t, kStep ks, lane, i): element pair K = 2i + 16*(lane>>4)
// + 32*ks of column n = 16*t + (lane&15), at offset ((t*ksteps+ks)*32+lane)*16 + 2i.
// ---------------------------------------------------------------------------
__global__ void pack_w_bf16(const float* __restrict__ W,
                            unsigned short* __restrict__ hi,
                            unsigned short* __restrict__ lo,
                            int K, int Cout)
{
    int tilesN = Cout >> 4, ksteps = K >> 5;
    int total = tilesN * ksteps * 32 * 8;
    int idx = blockIdx.x * blockDim.x + threadIdx.x;
    if (idx >= total) return;
    int i    = idx & 7;
    int r    = idx >> 3;
    int lane = r & 31;
    int r2   = r >> 5;
    int ks   = r2 % ksteps;
    int t    = r2 / ksteps;

    int kb = 2 * i + ((lane >> 4) << 4) + (ks << 5);
    int n  = (t << 4) + (lane & 15);
    float g0 = W[(size_t)kb * Cout + n];
    float g1 = W[(size_t)(kb + 1) * Cout + n];
    unsigned short h0 = f32_bf16_rne(g0), h1 = f32_bf16_rne(g1);
    size_t off = ((size_t)(t * ksteps + ks) * 32 + lane) * 16 + 2 * i;
    hi[off]     = h0;
    hi[off + 1] = h1;
    lo[off]     = f32_bf16_rne(g0 - bf16_hi_f32(h0));
    lo[off + 1] = f32_bf16_rne(g1 - bf16_hi_f32(h1));
}

// ---------------------------------------------------------------------------
// C[M x 16*TILES_N] = A[M x 32*KSTEPS] * W  with W pre-packed (hi/lo).
// One wave per 16-row tile computes ALL TILES_N column tiles: A converted once,
// B is pure b128 loads, 3 WMMAs per (tile, K-step) => near-fp32 via bf16 split.
// TILES_N/KSTEPS are compile-time so the tile loop is branch-free straight-line.
// A-operand lane order used here (16-bit A 16x32): M=lane&15; VGPR i (0..7) -> K=2i,2i+1
// (+16 for upper lane half), i.e. k = 16*half + element, same order as packed B.
// ---------------------------------------------------------------------------
template <int TILES_N, int KSTEPS>
__global__ void gat_gemm_wmma(const float* __restrict__ A,
                              const unsigned short* __restrict__ Bhi,
                              const unsigned short* __restrict__ Blo,
                              float* __restrict__ C,
                              int rowTiles)
{
    constexpr int Cout = TILES_N * 16;
    constexpr int K    = KSTEPS * 32;
    int gwave = (int)((blockIdx.x * blockDim.x + threadIdx.x) >> 5);
    if (gwave >= rowTiles) return;            // wave-uniform: EXEC stays all-1s
    int lane = threadIdx.x & 31;
    int half = lane >> 4;
    int l15  = lane & 15;
    int mBase = gwave << 4;

    v8f acc[TILES_N];
#pragma unroll
    for (int t = 0; t < TILES_N; ++t) acc[t] = (v8f){0.f,0.f,0.f,0.f,0.f,0.f,0.f,0.f};

    const float* arow = A + (size_t)(mBase + l15) * K;

    for (int ks = 0; ks < KSTEPS; ++ks) {
        BfVec aHi, aLo;
#pragma unroll
        for (int i = 0; i < 8; ++i) {
            int ka = (2 * i) + (half << 4) + (ks << 5);
            float2 f = *(const float2*)(arow + ka);
            unsigned short h0 = f32_bf16_rne(f.x), h1 = f32_bf16_rne(f.y);
            aHi.s[2*i]   = h0;
            aHi.s[2*i+1] = h1;
            aLo.s[2*i]   = f32_bf16_rne(f.x - bf16_hi_f32(h0));
            aLo.s[2*i+1] = f32_bf16_rne(f.y - bf16_hi_f32(h1));
        }
#pragma unroll
        for (int t = 0; t < TILES_N; ++t) {
            BfVec bHi, bLo;
            size_t boff = ((size_t)(t * KSTEPS + ks) * 32 + lane) * 16;
            bHi.q[0] = *(const uint4*)(Bhi + boff);
            bHi.q[1] = *(const uint4*)(Bhi + boff + 8);
            bLo.q[0] = *(const uint4*)(Blo + boff);
            bLo.q[1] = *(const uint4*)(Blo + boff + 8);
            acc[t] = __builtin_amdgcn_wmma_f32_16x16x32_bf16(false, aLo.v, false, bHi.v,
                                                             (short)0, acc[t], false, false);
            acc[t] = __builtin_amdgcn_wmma_f32_16x16x32_bf16(false, aHi.v, false, bLo.v,
                                                             (short)0, acc[t], false, false);
            acc[t] = __builtin_amdgcn_wmma_f32_16x16x32_bf16(false, aHi.v, false, bHi.v,
                                                             (short)0, acc[t], false, false);
        }
    }
    // C/D layout: VGPR r -> row = mBase + r + 8*half, col = 16*t + (lane&15)
    float* crow = C + (size_t)(mBase + (half << 3)) * Cout + l15;
#pragma unroll
    for (int t = 0; t < TILES_N; ++t) {
#pragma unroll
        for (int r = 0; r < 8; ++r)
            crow[(size_t)r * Cout + (t << 4)] = acc[t][r];
    }
}

// al_src[n] = xs[n,:].a_src ; al_dst[n] = xs[n,:].a_dst   (one wave per node)
__global__ void node_attn_logits(const float* __restrict__ xs,
                                 const float* __restrict__ a_src,
                                 const float* __restrict__ a_dst,
                                 float* __restrict__ alS, float* __restrict__ alD,
                                 int Nn, int C)
{
    int wave = (int)((blockIdx.x * blockDim.x + threadIdx.x) >> 5);
    int lane = threadIdx.x & 31;
    if (wave >= Nn) return;
    const float* row = xs + (size_t)wave * C;
    float s1 = 0.f, s2 = 0.f;
    for (int c = lane; c < C; c += 32) {
        float v = row[c];
        s1 += v * a_src[c];
        s2 += v * a_dst[c];
    }
    s1 = wave_sum(s1);
    s2 = wave_sum(s2);
    if (lane == 0) { alS[wave] = s1; alD[wave] = s2; }
}

// we[k] = We[k,:].a_edge  (EDGE_DIM=8 rows, one wave each; launch <<<1,256>>>)
__global__ void edge_weight_proj(const float* __restrict__ We,
                                 const float* __restrict__ a_edge,
                                 float* __restrict__ we, int C)
{
    int wave = threadIdx.x >> 5;
    int lane = threadIdx.x & 31;
    float s = 0.f;
    for (int c = lane; c < C; c += 32) s += We[(size_t)wave * C + c] * a_edge[c];
    s = wave_sum(s);
    if (lane == 0 && wave < EDGE_DIM) we[wave] = s;
}

__global__ void init_node_scalars(float* __restrict__ m, float* __restrict__ den, int Nn) {
    int i = blockIdx.x * blockDim.x + threadIdx.x;
    if (i < Nn) { m[i] = -__builtin_inff(); den[i] = 0.f; }
}

__global__ void init_out_bias(float* __restrict__ o, const float* __restrict__ bias,
                              int Nn, int C) {
    int total = Nn * C;
    for (int i = blockIdx.x * blockDim.x + threadIdx.x; i < total;
         i += gridDim.x * blockDim.x)
        o[i] = bias[i % C];
}

__global__ void zero_f(float* __restrict__ p, int n) {
    int i = blockIdx.x * blockDim.x + threadIdx.x;
    if (i < n) p[i] = 0.f;
}

// pass 1: alpha = leaky_relu(al_src[s] + al_dst[d] + edge_att.we); segment max
__global__ void edge_alpha(const int* __restrict__ src, const int* __restrict__ dst,
                           const float* __restrict__ eatt, const float* __restrict__ we,
                           const float* __restrict__ alS, const float* __restrict__ alD,
                           float* __restrict__ alpha, float* __restrict__ mmax, int Ecnt)
{
    int e = blockIdx.x * blockDim.x + threadIdx.x;
    if (e >= Ecnt) return;
    const float* ea = eatt + (size_t)e * EDGE_DIM;
    float ae = 0.f;
#pragma unroll
    for (int k = 0; k < EDGE_DIM; ++k) ae += ea[k] * we[k];
    int d = dst[e];
    float a = alS[src[e]] + alD[d] + ae;
    a = a > 0.f ? a : NEG_SLOPE * a;
    alpha[e] = a;
    atomicMaxF(&mmax[d], a);
}

// pass 2: ex = exp(alpha - m[dst]); segment sum
__global__ void edge_exp(const int* __restrict__ dst, const float* __restrict__ alpha,
                         const float* __restrict__ mmax, float* __restrict__ ex,
                         float* __restrict__ den, int Ecnt)
{
    int e = blockIdx.x * blockDim.x + threadIdx.x;
    if (e >= Ecnt) return;
    int d = dst[e];
    float v = __expf(alpha[e] - mmax[d]);
    ex[e] = v;
    atomicAdd(&den[d], v);
}

// pass 3: out[dst] += xs[src] * (ex/den[dst]); one wave per edge, float4/lane
__global__ void edge_scatter(const int* __restrict__ src, const int* __restrict__ dst,
                             const float* __restrict__ xs, const float* __restrict__ ex,
                             const float* __restrict__ den, float* __restrict__ out,
                             int Ecnt, int C)
{
    int gw = (int)((blockIdx.x * blockDim.x + threadIdx.x) >> 5);
    int lane = threadIdx.x & 31;
    if (gw >= Ecnt) return;
    int s = src[gw], d = dst[gw];
    float att = ex[gw] / den[d];
    int c = lane << 2;
    if (c < C) {
        float4 v = *(const float4*)(xs + (size_t)s * C + c);
        float* o = out + (size_t)d * C + c;
        atomicAdd(o + 0, v.x * att);
        atomicAdd(o + 1, v.y * att);
        atomicAdd(o + 2, v.z * att);
        atomicAdd(o + 3, v.w * att);
    }
}

// per-channel sum/sumsq; blockDim == C, register accumulate over rows
__global__ void bn_stats(const float* __restrict__ h, float* __restrict__ sum,
                         float* __restrict__ sumsq, int Nn, int C)
{
    int c = threadIdx.x;
    float s = 0.f, q = 0.f;
    for (int r = blockIdx.x; r < Nn; r += gridDim.x) {
        float v = h[(size_t)r * C + c];
        s += v;
        q += v * v;
    }
    atomicAdd(&sum[c], s);
    atomicAdd(&sumsq[c], q);
}

__global__ void bn_apply(const float* __restrict__ h, const float* __restrict__ sum,
                         const float* __restrict__ sumsq, const float* __restrict__ g,
                         const float* __restrict__ b, float* __restrict__ y,
                         int Nn, int C, int do_relu)
{
    int total = Nn * C;
    float invN = 1.f / (float)Nn;
    for (int i = blockIdx.x * blockDim.x + threadIdx.x; i < total;
         i += gridDim.x * blockDim.x) {
        int c = i % C;
        float mu  = sum[c] * invN;
        float var = sumsq[c] * invN - mu * mu;
        float v = g[c] * (h[i] - mu) * rsqrtf(var + BN_EPS) + b[c];
        if (do_relu) v = fmaxf(v, 0.f);
        y[i] = v;
    }
}

extern "C" void kernel_launch(void* const* d_in, const int* in_sizes, int n_in,
                              void* d_out, int out_size, void* d_ws, size_t ws_size,
                              hipStream_t stream)
{
    const float* x    = (const float*)d_in[0];
    const int*   ei   = (const int*)d_in[1];
    const float* eatt = (const float*)d_in[2];
    const float* W0   = (const float*)d_in[3];
    const float* aS0  = (const float*)d_in[4];
    const float* aD0  = (const float*)d_in[5];
    const float* We0  = (const float*)d_in[6];
    const float* aE0  = (const float*)d_in[7];
    const float* b0   = (const float*)d_in[8];
    const float* W1   = (const float*)d_in[9];
    const float* aS1  = (const float*)d_in[10];
    const float* aD1  = (const float*)d_in[11];
    const float* We1  = (const float*)d_in[12];
    const float* aE1  = (const float*)d_in[13];
    const float* b1   = (const float*)d_in[14];
    const float* bng  = (const float*)d_in[15];
    const float* bnb  = (const float*)d_in[16];
    const float* bfg  = (const float*)d_in[17];
    const float* bfb  = (const float*)d_in[18];
    const int* src = ei;
    const int* dst = ei + N_EDGES;
    float* outp = (float*)d_out;

    float* ws = (float*)d_ws;
    size_t o = 0;
    float* xs    = ws + o; o += (size_t)N_NODES * 128;   // projected features
    float* hbuf  = ws + o; o += (size_t)N_NODES * 128;   // scatter accum / hidden
    float* alpha = ws + o; o += (size_t)N_EDGES;
    float* exb   = ws + o; o += (size_t)N_EDGES;
    float* mmax  = ws + o; o += (size_t)N_NODES;
    float* den   = ws + o; o += (size_t)N_NODES;
    float* alS   = ws + o; o += (size_t)N_NODES;
    float* alD   = ws + o; o += (size_t)N_NODES;
    float* wev   = ws + o; o += 8;
    float* bnSum = ws + o; o += 128;
    float* bnSq  = ws + o; o += 128;
    unsigned short* wHi = (unsigned short*)(ws + o); o += 8192; // 8*4*32*16 shorts max
    unsigned short* wLo = (unsigned short*)(ws + o); o += 8192;
    (void)in_sizes; (void)n_in; (void)out_size; (void)ws_size;

    const int EB = (N_EDGES + 255) / 256;
    const int rowTiles = N_NODES / 16;                   // 3125

    // ---------------- Layer 0: IN_DIM(128) -> H_DIM(128) ----------------
    pack_w_bf16<<<32, 256, 0, stream>>>(W0, wHi, wLo, IN_DIM, H_DIM);
    gat_gemm_wmma<8, 4><<<(rowTiles + 7) / 8, 256, 0, stream>>>(x, wHi, wLo, xs, rowTiles);
    node_attn_logits<<<(N_NODES + 7) / 8, 256, 0, stream>>>(xs, aS0, aD0, alS, alD,
                                                            N_NODES, H_DIM);
    edge_weight_proj<<<1, 256, 0, stream>>>(We0, aE0, wev, H_DIM);
    init_node_scalars<<<(N_NODES + 255) / 256, 256, 0, stream>>>(mmax, den, N_NODES);
    init_out_bias<<<4096, 256, 0, stream>>>(hbuf, b0, N_NODES, H_DIM);
    zero_f<<<1, 256, 0, stream>>>(bnSum, 256); // zeroes bnSum + bnSq (contiguous)
    edge_alpha<<<EB, 256, 0, stream>>>(src, dst, eatt, wev, alS, alD, alpha, mmax, N_EDGES);
    edge_exp<<<EB, 256, 0, stream>>>(dst, alpha, mmax, exb, den, N_EDGES);
    edge_scatter<<<N_EDGES / 8, 256, 0, stream>>>(src, dst, xs, exb, den, hbuf,
                                                  N_EDGES, H_DIM);
    bn_stats<<<512, H_DIM, 0, stream>>>(hbuf, bnSum, bnSq, N_NODES, H_DIM);
    bn_apply<<<4096, 256, 0, stream>>>(hbuf, bnSum, bnSq, bng, bnb, hbuf,
                                       N_NODES, H_DIM, 1);

    // ---------------- Layer 1: H_DIM(128) -> OUT_DIM(112) ----------------
    pack_w_bf16<<<32, 256, 0, stream>>>(W1, wHi, wLo, H_DIM, OUT_DIM);
    gat_gemm_wmma<7, 4><<<(rowTiles + 7) / 8, 256, 0, stream>>>(hbuf, wHi, wLo, xs, rowTiles);
    node_attn_logits<<<(N_NODES + 7) / 8, 256, 0, stream>>>(xs, aS1, aD1, alS, alD,
                                                            N_NODES, OUT_DIM);
    edge_weight_proj<<<1, 256, 0, stream>>>(We1, aE1, wev, OUT_DIM);
    init_node_scalars<<<(N_NODES + 255) / 256, 256, 0, stream>>>(mmax, den, N_NODES);
    init_out_bias<<<4096, 256, 0, stream>>>(hbuf, b1, N_NODES, OUT_DIM);
    zero_f<<<1, 256, 0, stream>>>(bnSum, 256);
    edge_alpha<<<EB, 256, 0, stream>>>(src, dst, eatt, wev, alS, alD, alpha, mmax, N_EDGES);
    edge_exp<<<EB, 256, 0, stream>>>(dst, alpha, mmax, exb, den, N_EDGES);
    edge_scatter<<<N_EDGES / 8, 256, 0, stream>>>(src, dst, xs, exb, den, hbuf,
                                                  N_EDGES, OUT_DIM);
    bn_stats<<<512, OUT_DIM, 0, stream>>>(hbuf, bnSum, bnSq, N_NODES, OUT_DIM);
    bn_apply<<<4096, 256, 0, stream>>>(hbuf, bnSum, bnSq, bfg, bfb, outp,
                                       N_NODES, OUT_DIM, 0);
}
